// EdgeGCN_24927990186114
// MI455X (gfx1250) — compile-verified, matrix operand order checked by reading, not run on hardware
//
#include <hip/hip_runtime.h>
#include <hip/hip_bf16.h>

// ---------------------------------------------------------------------------
// EdgeGCN on gfx1250 (MI455X, wave32, WMMA).
// - All GEMM-shaped math: v_wmma_f32_16x16x32_f16 (f16 in, f32 accumulate).
// - Weights pre-swizzled into per-lane B-fragment layout; fragments addressed
//   as (per-lane base pointer + compile-time constant offset) so loads fold
//   into global_load_b128 with immediate IOFFSET.
// - Activations kept in f16 for A fragments -> contiguous b128 loads, no cvt.
// - Irregular segment-sum aggregation: global_atomic_add_f32 (h is L2-resident:
//   100k x 64 x 4B = 25.6 MB << 192 MB L2, so edge gathers are L2 hits).
// ---------------------------------------------------------------------------

typedef __attribute__((ext_vector_type(16))) _Float16 v16h;
typedef __attribute__((ext_vector_type(8)))  _Float16 h8;
typedef __attribute__((ext_vector_type(8)))  float    v8f;

#define HID 64

union V16 {
  v16h v;
  h8   h[2];
};

__device__ __forceinline__ v8f wmma_f32_f16(v16h a, v16h b, v8f c) {
  // 8 args: (neg_a, A, neg_b, B, c_mod, C, reuse_a, reuse_b)
  return __builtin_amdgcn_wmma_f32_16x16x32_f16(false, a, false, b, (short)0, c,
                                                false, false);
}

// ------------------------- elementwise helpers -----------------------------

__global__ void k_fill(float* p, float val, long n) {
  long i = (long)blockIdx.x * blockDim.x + threadIdx.x;
  long stride = (long)gridDim.x * blockDim.x;
  for (; i < n; i += stride) p[i] = val;
}

__global__ void k_deg_scatter(const int* __restrict__ v, float* __restrict__ deg, int E) {
  int i = blockIdx.x * blockDim.x + threadIdx.x;
  int stride = gridDim.x * blockDim.x;
  for (; i < E; i += stride) atomicAdd(&deg[v[i]], 1.0f);
}

__global__ void k_rsqrt_inplace(float* p, int n) {
  int i = blockIdx.x * blockDim.x + threadIdx.x;
  if (i < n) p[i] = rsqrtf(p[i]);
}

__global__ void k_edge_norm(const int* __restrict__ u, const int* __restrict__ v,
                            const float* __restrict__ dinv, float* __restrict__ norm, int E) {
  int i = blockIdx.x * blockDim.x + threadIdx.x;
  if (i < E) norm[i] = dinv[u[i]] * dinv[v[i]];
}

// dst[i] = (f16)src[i]
__global__ void k_cvt_f16(const float* __restrict__ src, _Float16* __restrict__ dst, long n) {
  long i = (long)blockIdx.x * blockDim.x + threadIdx.x;
  long stride = (long)gridDim.x * blockDim.x;
  for (; i < n; i += stride) dst[i] = (_Float16)src[i];
}

// Pre-swizzle a [rows x cols] f32 weight into WMMA B-fragment layout:
//   dst[((t*steps + s)*32 + lane)*16 + j], where lane half selects kb (0/8),
//   j<8 -> k = s*32+kb+j, j>=8 -> k = s*32+kb+8+j; n = t*16 + (lane&15).
// Rows beyond `rows` are zero (K padding, e.g. 136 -> 160).
__global__ void k_make_bfrag(const float* __restrict__ src, _Float16* __restrict__ dst,
                             int rows, int cols, int steps, int tiles) {
  int i = blockIdx.x * blockDim.x + threadIdx.x;
  int total = tiles * steps * 32 * 16;
  if (i >= total) return;
  int j    = i & 15;
  int fid  = i >> 4;
  int lane = fid & 31;
  int ts   = fid >> 5;
  int s    = ts % steps;
  int t    = ts / steps;
  int kb   = (lane >> 4) << 3;
  int n    = (t << 4) + (lane & 15);
  int k    = (s << 5) + kb + ((j < 8) ? j : (j + 8));
  float val = (k < rows) ? src[k * cols + n] : 0.0f;
  dst[i] = (_Float16)val;
}

// agg[v[e]] += hW[u[e]] * norm[e]   (float4 per thread, 16 threads per edge)
__global__ void k_scatter_agg(const int* __restrict__ u, const int* __restrict__ v,
                              const float* __restrict__ norm,
                              const float* __restrict__ hW,
                              float* __restrict__ agg, int E) {
  int i = blockIdx.x * blockDim.x + threadIdx.x;
  int total = E * 16;
  int stride = gridDim.x * blockDim.x;
  for (; i < total; i += stride) {
    int e  = i >> 4;
    int f4 = (i & 15) << 2;
    int ue = u[e], ve = v[e];
    float nm = norm[e];
    float4 val = *reinterpret_cast<const float4*>(hW + (long)ue * HID + f4);
    float* dst = agg + (long)ve * HID + f4;
    atomicAdd(dst + 0, val.x * nm);
    atomicAdd(dst + 1, val.y * nm);
    atomicAdd(dst + 2, val.z * nm);
    atomicAdd(dst + 3, val.w * nm);
  }
}

// hout16 = (f16) relu(agg + hW * dinv^2 + b)
__global__ void k_finalize(const float* __restrict__ agg, const float* __restrict__ hW,
                           const float* __restrict__ dinv, const float* __restrict__ b,
                           _Float16* __restrict__ hout16, int N) {
  long i = (long)blockIdx.x * blockDim.x + threadIdx.x;
  long total = (long)N * HID;
  if (i >= total) return;
  int node = (int)(i >> 6);
  int f = (int)(i & 63);
  float di = dinv[node];
  float x = agg[i] + hW[i] * di * di + b[f];
  hout16[i] = (_Float16)(x > 0.0f ? x : 0.0f);
}

// ------------------------- GCN GEMM: C[N,64] = A16[N,K] @ Bfrag ------------
// One wave owns 16 rows x 64 cols. K = STEPS*32, fully unrolled so fragment
// offsets are immediates relative to one per-lane base pointer.
template <int STEPS>
__global__ void k_gcn_gemm(const _Float16* __restrict__ A16,
                           const _Float16* __restrict__ Bf,
                           float* __restrict__ C, int N) {
  constexpr int K = STEPS * 32;
  int wave = (blockIdx.x * blockDim.x + threadIdx.x) >> 5;
  int lane = threadIdx.x & 31;
  int ntiles = (N + 15) >> 4;
  if (wave >= ntiles) return;            // wave-uniform exit, EXEC stays full
  int rowbase = wave << 4;
  int m  = lane & 15;
  int hi = lane >> 4;
  int kb = hi << 3;                      // 0 or 8
  int row = rowbase + m;
  if (row > N - 1) row = N - 1;          // clamp loads; stores handled below
  const _Float16* arow = A16 + (long)row * K + kb;  // per-lane A base
  const _Float16* bp   = Bf + lane * 16;            // per-lane fragment base

  v8f acc[4];
  #pragma unroll
  for (int t = 0; t < 4; ++t)
    #pragma unroll
    for (int j = 0; j < 8; ++j) acc[t][j] = 0.0f;

  #pragma unroll
  for (int s = 0; s < STEPS; ++s) {
    V16 a;
    a.h[0] = *(const h8*)(arow + s * 32);        // const-offset b128
    a.h[1] = *(const h8*)(arow + s * 32 + 16);
    v16h b0 = *(const v16h*)(bp + (0 * STEPS + s) * 512);
    v16h b1 = *(const v16h*)(bp + (1 * STEPS + s) * 512);
    v16h b2 = *(const v16h*)(bp + (2 * STEPS + s) * 512);
    v16h b3 = *(const v16h*)(bp + (3 * STEPS + s) * 512);
    acc[0] = wmma_f32_f16(a.v, b0, acc[0]);
    acc[1] = wmma_f32_f16(a.v, b1, acc[1]);
    acc[2] = wmma_f32_f16(a.v, b2, acc[2]);
    acc[3] = wmma_f32_f16(a.v, b3, acc[3]);
  }

  // C layout: VGPR r -> row (r + 8*hi), column (lane&15) + 16*t
  if (rowbase + 16 <= N) {               // full tile: no per-element guards
    #pragma unroll
    for (int t = 0; t < 4; ++t) {
      int n = (t << 4) + m;
      float* cp = C + (long)(rowbase + (hi << 3)) * HID + n;
      #pragma unroll
      for (int r = 0; r < 8; ++r) cp[(long)r * HID] = acc[t][r];
    }
  } else {
    #pragma unroll
    for (int t = 0; t < 4; ++t) {
      int n = (t << 4) + m;
      #pragma unroll
      for (int r = 0; r < 8; ++r) {
        int rr = rowbase + r + (hi << 3);
        if (rr < N) C[(long)rr * HID + n] = acc[t][r];
      }
    }
  }
}

// ------------------------- Edge MLP ----------------------------------------
// One wave per 16-edge tile.
//   z  [16 x 136->160] = [h[u] | h[v] | ef | 0-pad]  (built as f16 chunks)
//   a1 [16 x 64] = relu(z @ Wm1 + b1)    5 K-steps, 4 N-tiles  -> 20 WMMA
//   a2 [16 x 32] = relu(a1 @ Wm2 + b2)   2 K-steps, 2 N-tiles  ->  4 WMMA
//   out[16 x 1]  = a2 @ Wm3 + b3          VALU dot
#define MLP_WAVES 8

__global__ void k_edge_mlp(const _Float16* __restrict__ h16,
                           const int* __restrict__ u, const int* __restrict__ v,
                           const _Float16* __restrict__ ef16,
                           const _Float16* __restrict__ W1f,  // tiles=4, steps=5
                           const float* __restrict__ b1,
                           const _Float16* __restrict__ W2f,  // tiles=2, steps=2
                           const float* __restrict__ b2,
                           const float* __restrict__ W3,      // [32]
                           const float* __restrict__ b3,      // [1]
                           float* __restrict__ out, int E) {
  __shared__ __align__(16) _Float16 z1[MLP_WAVES][16][64];
  __shared__ float z2[MLP_WAVES][16][32];

  int wave = threadIdx.x >> 5;
  int lane = threadIdx.x & 31;
  long tile = (long)blockIdx.x * MLP_WAVES + wave;
  int m  = lane & 15;
  int hi = lane >> 4;
  int kb = hi << 3;

  long e  = tile * 16 + m;
  long ec = e < E ? e : (long)(E - 1);   // clamp loads; store is guarded
  int ue = u[ec], ve = v[ec];
  const _Float16* hu  = h16 + (long)ue * HID + kb;   // per-lane A bases
  const _Float16* hv  = h16 + (long)ve * HID + kb;
  const _Float16* efe = ef16 + ec * 8;
  const _Float16* w1p = W1f + lane * 16;             // per-lane fragment bases
  const _Float16* w2p = W2f + lane * 16;

  h8 zero8;
  #pragma unroll
  for (int j = 0; j < 8; ++j) zero8[j] = (_Float16)0.0f;

  // ---- layer 1: 16x160 @ 160x64 ----
  v8f acc[4];
  #pragma unroll
  for (int t = 0; t < 4; ++t)
    #pragma unroll
    for (int j = 0; j < 8; ++j) acc[t][j] = 0.0f;

  #pragma unroll
  for (int s = 0; s < 5; ++s) {
    V16 a;
    if (s < 2) {                         // K 0..63 : h[u]
      a.h[0] = *(const h8*)(hu + (s << 5));
      a.h[1] = *(const h8*)(hu + (s << 5) + 16);
    } else if (s < 4) {                  // K 64..127 : h[v]
      a.h[0] = *(const h8*)(hv + ((s - 2) << 5));
      a.h[1] = *(const h8*)(hv + ((s - 2) << 5) + 16);
    } else {                             // K 128..159 : ef | zero pad
      a.h[0] = (kb == 0) ? *(const h8*)(efe) : zero8;
      a.h[1] = zero8;
    }
    v16h b0 = *(const v16h*)(w1p + (0 * 5 + s) * 512);
    v16h b1 = *(const v16h*)(w1p + (1 * 5 + s) * 512);
    v16h b2 = *(const v16h*)(w1p + (2 * 5 + s) * 512);
    v16h b3 = *(const v16h*)(w1p + (3 * 5 + s) * 512);
    acc[0] = wmma_f32_f16(a.v, b0, acc[0]);
    acc[1] = wmma_f32_f16(a.v, b1, acc[1]);
    acc[2] = wmma_f32_f16(a.v, b2, acc[2]);
    acc[3] = wmma_f32_f16(a.v, b3, acc[3]);
  }
  // bias + relu, transpose C-layout -> row-major f16 LDS (A-layout source)
  #pragma unroll
  for (int t = 0; t < 4; ++t) {
    int n = (t << 4) + m;
    float bb = b1[n];
    #pragma unroll
    for (int r = 0; r < 8; ++r) {
      float x = acc[t][r] + bb;
      z1[wave][r + (hi << 3)][n] = (_Float16)(x > 0.0f ? x : 0.0f);
    }
  }
  __syncthreads();

  // ---- layer 2: 16x64 @ 64x32 ----
  v8f acc2[2];
  #pragma unroll
  for (int t = 0; t < 2; ++t)
    #pragma unroll
    for (int j = 0; j < 8; ++j) acc2[t][j] = 0.0f;

  const _Float16* z1p = &z1[wave][m][kb];
  #pragma unroll
  for (int s = 0; s < 2; ++s) {
    V16 a;
    a.h[0] = *(const h8*)(z1p + (s << 5));          // ds_load b128
    a.h[1] = *(const h8*)(z1p + (s << 5) + 16);
    v16h b0 = *(const v16h*)(w2p + (0 * 2 + s) * 512);
    v16h b1 = *(const v16h*)(w2p + (1 * 2 + s) * 512);
    acc2[0] = wmma_f32_f16(a.v, b0, acc2[0]);
    acc2[1] = wmma_f32_f16(a.v, b1, acc2[1]);
  }
  #pragma unroll
  for (int t = 0; t < 2; ++t) {
    int n = (t << 4) + m;
    float bb = b2[n];
    #pragma unroll
    for (int r = 0; r < 8; ++r) {
      float x = acc2[t][r] + bb;
      z2[wave][r + (hi << 3)][n] = x > 0.0f ? x : 0.0f;
    }
  }
  __syncthreads();

  // ---- layer 3: 16x32 @ 32x1 (VALU dot) ----
  if (lane < 16) {
    long eo = tile * 16 + lane;
    if (eo < E) {
      float s = b3[0];
      #pragma unroll
      for (int k = 0; k < 32; ++k) s += z2[wave][lane][k] * W3[k];
      out[eo] = s;
    }
  }
}

// ------------------------- host launch -------------------------------------

static inline size_t align256(size_t x) { return (x + 255) & ~(size_t)255; }

extern "C" void kernel_launch(void* const* d_in, const int* in_sizes, int n_in,
                              void* d_out, int out_size, void* d_ws, size_t ws_size,
                              hipStream_t stream) {
  const float* x   = (const float*)d_in[0];
  const int*   ei  = (const int*)d_in[1];   // [2, E] int32
  const float* ef  = (const float*)d_in[2];
  const float* W0  = (const float*)d_in[3];
  const float* b0  = (const float*)d_in[4];
  const float* W1  = (const float*)d_in[5];
  const float* b1  = (const float*)d_in[6];
  const float* W2  = (const float*)d_in[7];
  const float* b2  = (const float*)d_in[8];
  const float* Wm1 = (const float*)d_in[9];
  const float* bm1 = (const float*)d_in[10];
  const float* Wm2 = (const float*)d_in[11];
  const float* bm2 = (const float*)d_in[12];
  const float* Wm3 = (const float*)d_in[13];
  const float* bm3 = (const float*)d_in[14];
  float* out = (float*)d_out;

  const int N = in_sizes[0] / 32;
  const int E = in_sizes[2] / 8;
  const int* u = ei;
  const int* v = ei + E;

  // workspace carve-out
  char* ws = (char*)d_ws;
  float* deg = (float*)ws;      ws += align256(sizeof(float) * (size_t)N);
  float* nrm = (float*)ws;      ws += align256(sizeof(float) * (size_t)E);
  float* hW  = (float*)ws;      ws += align256(sizeof(float) * (size_t)N * HID);
  float* agg = (float*)ws;      ws += align256(sizeof(float) * (size_t)N * HID);
  _Float16* x16  = (_Float16*)ws; ws += align256(sizeof(_Float16) * (size_t)N * 32);
  _Float16* hA16 = (_Float16*)ws; ws += align256(sizeof(_Float16) * (size_t)N * HID);
  _Float16* ef16 = (_Float16*)ws; ws += align256(sizeof(_Float16) * (size_t)E * 8);
  _Float16* W0f  = (_Float16*)ws; ws += align256(sizeof(_Float16) * 4 * 1 * 512);
  _Float16* W1f  = (_Float16*)ws; ws += align256(sizeof(_Float16) * 4 * 2 * 512);
  _Float16* W2f  = (_Float16*)ws; ws += align256(sizeof(_Float16) * 4 * 2 * 512);
  _Float16* Wm1f = (_Float16*)ws; ws += align256(sizeof(_Float16) * 4 * 5 * 512);
  _Float16* Wm2f = (_Float16*)ws; ws += align256(sizeof(_Float16) * 2 * 2 * 512);

  const int TPB = 256;
  auto blocks = [](long n, int t) { return (int)((n + t - 1) / t); };

  // degrees (+self-loop) -> dinv -> per-edge norm
  k_fill<<<blocks(N, TPB), TPB, 0, stream>>>(deg, 1.0f, N);
  k_deg_scatter<<<blocks(E, TPB), TPB, 0, stream>>>(v, deg, E);
  k_rsqrt_inplace<<<blocks(N, TPB), TPB, 0, stream>>>(deg, N);   // deg -> dinv
  k_edge_norm<<<blocks(E, TPB), TPB, 0, stream>>>(u, v, deg, nrm, E);

  // f16 activations and pre-swizzled weight fragments
  k_cvt_f16<<<blocks((long)N * 32, TPB), TPB, 0, stream>>>(x, x16, (long)N * 32);
  k_cvt_f16<<<blocks((long)E * 8, TPB), TPB, 0, stream>>>(ef, ef16, (long)E * 8);
  k_make_bfrag<<<blocks(4 * 1 * 512, TPB), TPB, 0, stream>>>(W0,  W0f,  32,  64, 1, 4);
  k_make_bfrag<<<blocks(4 * 2 * 512, TPB), TPB, 0, stream>>>(W1,  W1f,  64,  64, 2, 4);
  k_make_bfrag<<<blocks(4 * 2 * 512, TPB), TPB, 0, stream>>>(W2,  W2f,  64,  64, 2, 4);
  k_make_bfrag<<<blocks(4 * 5 * 512, TPB), TPB, 0, stream>>>(Wm1, Wm1f, 136, 64, 5, 4);
  k_make_bfrag<<<blocks(2 * 2 * 512, TPB), TPB, 0, stream>>>(Wm2, Wm2f, 64,  32, 2, 2);

  const int gemm_blocks = blocks(((long)N + 15) / 16, TPB / 32);
  const long nh = (long)N * HID;
  const int sc_blocks = blocks((long)E * 16, TPB);

  // layer 0: x[N,32] @ W0
  k_gcn_gemm<1><<<gemm_blocks, TPB, 0, stream>>>(x16, W0f, hW, N);
  k_fill<<<blocks(nh, TPB), TPB, 0, stream>>>(agg, 0.0f, nh);
  k_scatter_agg<<<sc_blocks, TPB, 0, stream>>>(u, v, nrm, hW, agg, E);
  k_finalize<<<blocks(nh, TPB), TPB, 0, stream>>>(agg, hW, deg, b0, hA16, N);

  // layer 1
  k_gcn_gemm<2><<<gemm_blocks, TPB, 0, stream>>>(hA16, W1f, hW, N);
  k_fill<<<blocks(nh, TPB), TPB, 0, stream>>>(agg, 0.0f, nh);
  k_scatter_agg<<<sc_blocks, TPB, 0, stream>>>(u, v, nrm, hW, agg, E);
  k_finalize<<<blocks(nh, TPB), TPB, 0, stream>>>(agg, hW, deg, b1, hA16, N);

  // layer 2
  k_gcn_gemm<2><<<gemm_blocks, TPB, 0, stream>>>(hA16, W2f, hW, N);
  k_fill<<<blocks(nh, TPB), TPB, 0, stream>>>(agg, 0.0f, nh);
  k_scatter_agg<<<sc_blocks, TPB, 0, stream>>>(u, v, nrm, hW, agg, E);
  k_finalize<<<blocks(nh, TPB), TPB, 0, stream>>>(agg, hW, deg, b2, hA16, N);

  // edge MLP (one wave per 16 edges)
  long tiles = ((long)E + 15) / 16;
  int mlp_blocks = (int)((tiles + MLP_WAVES - 1) / MLP_WAVES);
  k_edge_mlp<<<mlp_blocks, MLP_WAVES * 32, 0, stream>>>(
      hA16, u, v, ef16, Wm1f, bm1, Wm2f, bm2, Wm3, bm3, out, E);
}
